// GNNModule_33165737459916
// MI455X (gfx1250) — compile-verified, matrix operand order checked by reading, not run on hardware
//
#include <hip/hip_runtime.h>

typedef __attribute__((ext_vector_type(16))) __bf16 v16bf;
typedef __attribute__((ext_vector_type(8)))  __bf16 v8bf;
typedef __attribute__((ext_vector_type(8)))  float  v8f;

// gfx1250 async global->LDS path if this toolchain exposes it (guarded so the
// fallback — plain LDS stores — keeps the file compiling on any toolchain).
#if defined(__has_builtin)
#if __has_builtin(__builtin_amdgcn_global_load_async_to_lds_b128) && \
    __has_builtin(__builtin_amdgcn_s_wait_asynccnt)
#define USE_ASYNC_LDS 1
#endif
#endif
#ifndef USE_ASYNC_LDS
#define USE_ASYNC_LDS 0
#endif

#define AS_LDS __attribute__((address_space(3)))
typedef int async_v4i __attribute__((vector_size(4 * sizeof(int))));

// ---------------------------------------------------------------- utilities

__global__ void fill_f32(float* __restrict__ p, float v, long long n) {
    long long i = (long long)blockIdx.x * blockDim.x + threadIdx.x;
    if (i < n) p[i] = v;
}

// Wt[n*K + k] = (bf16) W[k*N + n]   (transpose + downconvert weight matrix)
__global__ void transpose_to_bf16(const float* __restrict__ W,
                                  __bf16* __restrict__ Wt, int K, int N) {
    int i = blockIdx.x * blockDim.x + threadIdx.x;
    if (i >= K * N) return;
    int n = i / K, k = i % K;
    Wt[i] = (__bf16)W[(size_t)k * N + n];
}

__global__ void deg_count(const int* __restrict__ dst, float* __restrict__ deg, int E) {
    int e = blockIdx.x * blockDim.x + threadIdx.x;
    if (e < E) unsafeAtomicAdd(&deg[dst[e]], 1.0f);
}

__global__ void rsqrt_inplace(float* __restrict__ d, int N) {
    int i = blockIdx.x * blockDim.x + threadIdx.x;
    if (i < N) d[i] = rsqrtf(d[i]);   // deg >= 1 always (self loop)
}

// ---------------------------------------------------------------- WMMA GEMM
// H[M x (NT*16)] = act(X)[M x K] * W[K x (NT*16)], W given transposed as
// Wt[(NT*16) x K] bf16.  8 waves/block; one wave per 16-row tile; NT 16x16
// f32 accumulators spanning the full output width; K stepped by 32.
//
// CDNA5 16-bit A layout: lane L holds row M = L%16; with h = L/16, VGPRs
// 0..3 hold K = 8h..8h+7 and VGPRs 4..7 hold K = 16+8h..16+8h+7.  B (32x16)
// mirrors this with N = L%16.  Both map to contiguous 8-element chunks.
//
// Wt is staged once per block into LDS (row stride padded by 8 bf16 = 4
// dwords, so the 16 lanes of each half-wave hit disjoint 4-bank groups:
// conflict-free ds_load_b128).  FUSE_BIAS_RELU applies relu(x + bias[k]) to
// the A operand during the fp32->bf16 conversion (layer-2 fusion).
template <int NT, int K, bool FUSE_BIAS_RELU>
__global__ __launch_bounds__(256) void
gemm_xw_bf16(const float* __restrict__ X, const __bf16* __restrict__ Wt,
             const float* __restrict__ bias, float* __restrict__ H, int M) {
    constexpr int N  = NT * 16;
    constexpr int KP = K + 8;                     // padded LDS row stride
    __shared__ __bf16 sB[N * KP];

    // ---- cooperative stage: Wt (N x K bf16) -> LDS in 16B chunks
    {
        constexpr int CHUNKS = N * (K / 8);
        for (int c = threadIdx.x; c < CHUNKS; c += 256) {
            const int n  = c / (K / 8);
            const int kc = c - n * (K / 8);
            const __bf16* src = Wt + (size_t)n * K + kc * 8;
            __bf16*       dst = sB + n * KP + kc * 8;
#if USE_ASYNC_LDS
            __builtin_amdgcn_global_load_async_to_lds_b128(
                (async_v4i*)src,
                (AS_LDS async_v4i*)(unsigned)(unsigned long long)dst, 0, 0);
#else
            *(uint4*)dst = *(const uint4*)src;
#endif
        }
#if USE_ASYNC_LDS
        __builtin_amdgcn_s_wait_asynccnt(0);
#endif
        __syncthreads();                          // before any wave can exit
    }

    const int wave    = threadIdx.x >> 5;
    const int lane    = threadIdx.x & 31;
    const int rowTile = blockIdx.x * 8 + wave;
    if (rowTile * 16 >= M) return;                // M is a multiple of 16

    const int mn   = lane & 15;                   // A-row / B-col of this lane
    const int half = lane >> 4;                   // K-half select
    const float* __restrict__ xrow = X + (size_t)(rowTile * 16 + mn) * K;

    v8f acc[NT];
    const v8f vzero = {0.f, 0.f, 0.f, 0.f, 0.f, 0.f, 0.f, 0.f};
#pragma unroll
    for (int t = 0; t < NT; ++t) acc[t] = vzero;

    for (int kk = 0; kk < K; kk += 32) {
        // ---- A fragment: two 32B fp32 chunks -> bf16 in regs
        float4 a0 = *(const float4*)(xrow + kk + 8 * half);
        float4 a1 = *(const float4*)(xrow + kk + 8 * half + 4);
        float4 a2 = *(const float4*)(xrow + kk + 16 + 8 * half);
        float4 a3 = *(const float4*)(xrow + kk + 16 + 8 * half + 4);
        if constexpr (FUSE_BIAS_RELU) {
            auto br4 = [](float4 a, float4 b) {
                float4 r;
                r.x = fmaxf(a.x + b.x, 0.f);
                r.y = fmaxf(a.y + b.y, 0.f);
                r.z = fmaxf(a.z + b.z, 0.f);
                r.w = fmaxf(a.w + b.w, 0.f);
                return r;
            };
            a0 = br4(a0, *(const float4*)(bias + kk + 8 * half));
            a1 = br4(a1, *(const float4*)(bias + kk + 8 * half + 4));
            a2 = br4(a2, *(const float4*)(bias + kk + 16 + 8 * half));
            a3 = br4(a3, *(const float4*)(bias + kk + 16 + 8 * half + 4));
        }
        v16bf a;
        a[0]  = (__bf16)a0.x;  a[1]  = (__bf16)a0.y;
        a[2]  = (__bf16)a0.z;  a[3]  = (__bf16)a0.w;
        a[4]  = (__bf16)a1.x;  a[5]  = (__bf16)a1.y;
        a[6]  = (__bf16)a1.z;  a[7]  = (__bf16)a1.w;
        a[8]  = (__bf16)a2.x;  a[9]  = (__bf16)a2.y;
        a[10] = (__bf16)a2.z;  a[11] = (__bf16)a2.w;
        a[12] = (__bf16)a3.x;  a[13] = (__bf16)a3.y;
        a[14] = (__bf16)a3.z;  a[15] = (__bf16)a3.w;

#pragma unroll
        for (int t = 0; t < NT; ++t) {
            const __bf16* wrow = sB + (t * 16 + mn) * KP;
            v8bf blo = *(const v8bf*)(wrow + kk + 8 * half);       // ds b128
            v8bf bhi = *(const v8bf*)(wrow + kk + 16 + 8 * half);  // ds b128
            v16bf b = __builtin_shufflevector(blo, bhi,
                        0, 1, 2, 3, 4, 5, 6, 7, 8, 9, 10, 11, 12, 13, 14, 15);
            acc[t] = __builtin_amdgcn_wmma_f32_16x16x32_bf16(
                         /*neg_a=*/false, a, /*neg_b=*/false, b,
                         /*c_mod=*/(short)0, acc[t],
                         /*reuse_a=*/false, /*reuse_b=*/false);
        }
    }

    // D layout: VGPR r -> M = 8*half + r, N = lane%16
#pragma unroll
    for (int t = 0; t < NT; ++t) {
#pragma unroll
        for (int r = 0; r < 8; ++r) {
            int row = rowTile * 16 + half * 8 + r;
            H[(size_t)row * N + t * 16 + mn] = acc[t][r];
        }
    }
}

// ---------------------------------------------------------------- scatter

// agg[dst] += dinv[src]*dinv[dst] * h[src]  (1 thread per edge x 4 features;
// with F=128 a wave maps to exactly one edge -> uniform index loads)
__global__ void edge_aggregate(const float* __restrict__ h,
                               const int* __restrict__ src,
                               const int* __restrict__ dst,
                               const float* __restrict__ dinv,
                               float* __restrict__ agg, long long E, int F) {
    const int Fc = F >> 2;
    long long tid = (long long)blockIdx.x * blockDim.x + threadIdx.x;
    if (tid >= E * Fc) return;
    int e = (int)(tid / Fc);
    int c = (int)(tid % Fc);
    int s = src[e], d = dst[e];
    float nrm = dinv[s] * dinv[d];
    float4 v = *(const float4*)(h + (size_t)s * F + c * 4);
    float* o = agg + (size_t)d * F + c * 4;
    unsafeAtomicAdd(o + 0, nrm * v.x);
    unsafeAtomicAdd(o + 1, nrm * v.y);
    unsafeAtomicAdd(o + 2, nrm * v.z);
    unsafeAtomicAdd(o + 3, nrm * v.w);
}

// self loop: agg[i] += dinv[node]^2 * h[i]  (element-exclusive, no atomics)
__global__ void self_loop_add(const float* __restrict__ h,
                              const float* __restrict__ dinv,
                              float* __restrict__ agg, long long n, int F) {
    long long i = (long long)blockIdx.x * blockDim.x + threadIdx.x;
    if (i >= n) return;
    int node = (int)(i / F);
    float dv = dinv[node];
    agg[i] += dv * dv * h[i];
}

__global__ void bias_add(float* __restrict__ a, const float* __restrict__ b,
                         long long n, int F) {
    long long i = (long long)blockIdx.x * blockDim.x + threadIdx.x;
    if (i >= n) return;
    a[i] += b[(int)(i % F)];
}

// ---------------------------------------------------------------- driver

extern "C" void kernel_launch(void* const* d_in, const int* in_sizes, int n_in,
                              void* d_out, int out_size, void* d_ws, size_t ws_size,
                              hipStream_t stream) {
    const float* x  = (const float*)d_in[0];
    const int*   ei = (const int*)d_in[1];   // (2, E) int32
    const float* W1 = (const float*)d_in[2];
    const float* b1 = (const float*)d_in[3];
    const float* W2 = (const float*)d_in[4];
    const float* b2 = (const float*)d_in[5];

    const int F_HID = in_sizes[3];                       // 128
    const int F_OUT = in_sizes[5];                       // 64
    const int F_IN  = in_sizes[2] / F_HID;               // 256
    const long long N = (long long)in_sizes[0] / F_IN;   // 100000
    const long long E = (long long)in_sizes[1] / 2;      // 3200000
    const int* src = ei;
    const int* dst = ei + E;

    // workspace carve-up (256B-aligned slices)
    char*  ws  = (char*)d_ws;
    size_t off = 0;
    auto carve = [&](size_t bytes) -> void* {
        void* p = ws + off;
        off = (off + bytes + 255) & ~(size_t)255;
        return p;
    };
    __bf16* Wt1  = (__bf16*)carve((size_t)F_HID * F_IN * sizeof(__bf16));
    __bf16* Wt2  = (__bf16*)carve((size_t)F_OUT * F_HID * sizeof(__bf16));
    float*  dinv = (float*)carve((size_t)N * sizeof(float));   // deg -> dinv
    float*  h1   = (float*)carve((size_t)N * F_HID * sizeof(float));
    float*  agg1 = (float*)carve((size_t)N * F_HID * sizeof(float));
    float*  h2   = (float*)carve((size_t)N * F_OUT * sizeof(float));
    if (off > ws_size) return;                 // ws too small: bail (no UB)
    float* out = (float*)d_out;

    const int T = 256;
    auto nb = [](long long n, int t) { return (unsigned)((n + t - 1) / t); };

    // init: deg = 1 (self loops), zero accumulators/output (d_out is poisoned)
    fill_f32<<<nb(N, T), T, 0, stream>>>(dinv, 1.0f, N);
    fill_f32<<<nb(N * F_HID, T), T, 0, stream>>>(agg1, 0.0f, N * F_HID);
    fill_f32<<<nb(N * F_OUT, T), T, 0, stream>>>(out, 0.0f, N * F_OUT);

    // weights -> transposed bf16
    transpose_to_bf16<<<nb((long long)F_IN * F_HID, T), T, 0, stream>>>(W1, Wt1, F_IN, F_HID);
    transpose_to_bf16<<<nb((long long)F_HID * F_OUT, T), T, 0, stream>>>(W2, Wt2, F_HID, F_OUT);

    // symmetric norm: deg -> D^{-1/2}
    deg_count<<<nb(E, T), T, 0, stream>>>(dst, dinv, (int)E);
    rsqrt_inplace<<<nb(N, T), T, 0, stream>>>(dinv, (int)N);

    const long long tiles = (N + 15) / 16;
    const unsigned  gblks = (unsigned)((tiles + 7) / 8);   // 8 waves / block

    // ---- layer 1: h1 = X @ W1 (WMMA) ; agg1 = norm-scatter(h1)
    if (F_IN == 256 && F_HID == 128)
        gemm_xw_bf16<8, 256, false><<<gblks, 256, 0, stream>>>(x, Wt1, nullptr, h1, (int)N);
    edge_aggregate<<<nb(E * (F_HID / 4), T), T, 0, stream>>>(h1, src, dst, dinv, agg1, E, F_HID);
    self_loop_add<<<nb(N * F_HID, T), T, 0, stream>>>(h1, dinv, agg1, N * F_HID, F_HID);

    // ---- layer 2: h2 = relu(agg1 + b1) @ W2 (bias+ReLU fused into A-load)
    if (F_HID == 128 && F_OUT == 64)
        gemm_xw_bf16<4, 128, true><<<gblks, 256, 0, stream>>>(agg1, Wt2, b1, h2, (int)N);
    edge_aggregate<<<nb(E * (F_OUT / 4), T), T, 0, stream>>>(h2, src, dst, dinv, out, E, F_OUT);
    self_loop_add<<<nb(N * F_OUT, T), T, 0, stream>>>(h2, dinv, out, N * F_OUT, F_OUT);
    bias_add<<<nb(N * F_OUT, T), T, 0, stream>>>(out, b2, N * F_OUT, F_OUT);
}
